// Generator_89120571392390
// MI455X (gfx1250) — matrix operand order repaired and checked by reading, not compiled, not run on hardware
//
#include <hip/hip_runtime.h>
#include <hip/hip_bf16.h>

// ---------------------------------------------------------------------------
// Persistent-kernel GRU generator for MI455X (gfx1250, wave32, WMMA).
//   B=512 batch, H=512 hidden, IN=128 (= 1 + 127), L=seq_len (256).
// Per step:  s0 = GRU1([x,y], s0) ; s1 = GRU2(s0, s1) ; x = relu(fc(s1))
// gi1 = giy + x*w_ih1[:,0]  (giy precomputed once: y @ w_ih1[:,1:]^T + b_ih1)
// All GEMMs: bf16 inputs, f32 WMMA accumulation, f32 master state.
// ---------------------------------------------------------------------------

constexpr int kB   = 512;
constexpr int kH   = 512;
constexpr int kIN  = 128;
constexpr int kYD  = 127;
constexpr int kZD  = 385;
constexpr int kG3  = 3 * kH;      // 1536 gate rows
constexpr int kNWG = 64;          // persistent workgroups (must all be resident)
constexpr int kTPB = 256;         // 8 wave32 per WG -> 512 waves total

typedef __attribute__((ext_vector_type(16))) __bf16 v16bf;
typedef __attribute__((ext_vector_type(8)))  float  v8f;

union Frag32B { v16bf v; uint4 q[2]; };

__device__ __forceinline__ unsigned short f2bf(float f) {
  union { float f; unsigned int u; } x; x.f = f;
  unsigned int r = x.u + 0x7FFFu + ((x.u >> 16) & 1u);   // round-to-nearest-even
  return (unsigned short)(r >> 16);
}
__device__ __forceinline__ float sigmoid_f(float x) { return 1.0f / (1.0f + __expf(-x)); }

// A-fragment: 16x32 bf16 tile, rows m0..m0+15, cols k0..k0+31 of row-major [.,ld].
// CDNA5 layout: lane<16 holds K={0..7,16..23}, lane>=16 holds K={8..15,24..31}.
__device__ __forceinline__ v16bf load_a_frag(const unsigned short* base, int row,
                                             int ld, int k0, int lane) {
  const unsigned short* p = base + row * ld + k0 + ((lane >> 4) << 3);
  Frag32B f;
  f.q[0] = *(const uint4*)(p);
  f.q[1] = *(const uint4*)(p + 16);
  return f.v;
}

// B-fragment: 32x16 bf16 tile = transpose of 16 rows of row-major W[o, k].
// lane<16: column o0+lane K={0..7,16..23}; lane>=16: K={8..15,24..31}.
__device__ __forceinline__ v16bf load_b_frag(const unsigned short* base, int orow,
                                             int ld, int k0, int lane) {
  const unsigned short* p = base + orow * ld + k0 + ((lane >> 4) << 3);
  Frag32B f;
  f.q[0] = *(const uint4*)(p);
  f.q[1] = *(const uint4*)(p + 16);
  return f.v;
}

__device__ __forceinline__ v8f wmma_bf16(v16bf a, v16bf b, v8f c) {
  return __builtin_amdgcn_wmma_f32_16x16x32_bf16(false, a, false, b,
                                                 (short)0, c, false, false);
}

// Monotonic device-wide barrier: counter never resets within a launch.
__device__ __forceinline__ void gbar(int* cnt, int& epoch) {
  __threadfence();
  __syncthreads();
  if (threadIdx.x == 0) {
    __hip_atomic_fetch_add(cnt, 1, __ATOMIC_RELEASE, __HIP_MEMORY_SCOPE_AGENT);
    const int target = (epoch + 1) * kNWG;
    while (__hip_atomic_load(cnt, __ATOMIC_ACQUIRE, __HIP_MEMORY_SCOPE_AGENT) < target)
      __builtin_amdgcn_s_sleep(2);
  }
  epoch++;
  __syncthreads();
  __threadfence();
}

__global__ void init_barrier_kernel(int* bar) { *bar = 0; }

__global__ void __launch_bounds__(kTPB, 1)
gru_persistent(const float* __restrict__ z1,     const float* __restrict__ yin,
               const float* __restrict__ x0,     const float* __restrict__ hidden1,
               const float* __restrict__ w_ih1,  const float* __restrict__ w_hh1,
               const float* __restrict__ b_ih1,  const float* __restrict__ b_hh1,
               const float* __restrict__ w_ih2,  const float* __restrict__ w_hh2,
               const float* __restrict__ b_ih2,  const float* __restrict__ b_hh2,
               const float* __restrict__ w_fc,   const float* __restrict__ b_fc,
               const int*   __restrict__ seq_len_p,
               float* __restrict__ out,
               int*   __restrict__ bar,
               float* __restrict__ xcur,
               float* __restrict__ s0f, float* __restrict__ s1f,
               unsigned short* __restrict__ s0b, unsigned short* __restrict__ s1b,
               float* __restrict__ giy,
               unsigned short* __restrict__ whh1b,
               unsigned short* __restrict__ wih2b,
               unsigned short* __restrict__ whh2b,
               float* __restrict__ wcol0)
{
  const int tid  = threadIdx.x;
  const int lane = tid & 31;
  const int gw   = blockIdx.x * (kTPB / 32) + (tid >> 5);   // 0..511
  const int gtid = blockIdx.x * kTPB + tid;
  const int nthr = kNWG * kTPB;                             // 16384
  int ep = 0;

  const int L = *seq_len_p;

  // ---------------- PREP (once per launch; deterministic) ----------------
  for (int i = gtid; i < kG3 * kH; i += nthr) {
    whh1b[i] = f2bf(w_hh1[i]);
    wih2b[i] = f2bf(w_ih2[i]);
    whh2b[i] = f2bf(w_hh2[i]);
  }
  for (int g = gtid; g < kG3; g += nthr) wcol0[g] = w_ih1[g * kIN];
  for (int i = gtid; i < kB * kH; i += nthr) {
    const int b = i >> 9, h = i & (kH - 1);
    const float v0 = (h < kZD) ? z1[b * kZD + h] : yin[b * kYD + (h - kZD)];
    s0f[i] = v0;            s0b[i] = f2bf(v0);
    const float v1 = hidden1[i];
    s1f[i] = v1;            s1b[i] = f2bf(v1);
  }
  for (int b = gtid; b < kB; b += nthr) xcur[b] = x0[b];
  for (int p = gtid; p < kB * kG3; p += nthr) {          // giy = y @ w_ih1[:,1:]^T + b_ih1
    const int b = p / kG3, g = p - b * kG3;
    float acc = b_ih1[g];
    const float* wr = w_ih1 + g * kIN + 1;
    const float* yr = yin + b * kYD;
    for (int j = 0; j < kYD; ++j) acc = fmaf(yr[j], wr[j], acc);
    giy[p] = acc;
  }
  gbar(bar, ep);

  // Wave tiling: ntile = fixed 16 hidden columns; 2 batch tiles per wave.
  const int ntile = gw & 31;
  const int mgrp  = gw >> 5;                 // 0..15
  const int h0    = ntile * 16;
  const int hcol  = h0 + (lane & 15);
  const int bhalf = (lane >> 4) << 3;        // 0 or 8 (C/D row offset)

  // ---------------- SEQUENCE LOOP ----------------
  for (int t = 0; t < L; ++t) {
    const int inb  = t & 1;
    const int outb = inb ^ 1;
    const unsigned short* s0_in  = s0b + inb  * (kB * kH);
    const unsigned short* s0_new = s0b + outb * (kB * kH);
    const unsigned short* s1_in  = s1b + inb  * (kB * kH);

    // ---- PHASE 1: s0_new = GRU1([x,y], s0) ; only w_hh1 GEMM + rank-1 gi ----
    {
      v8f acc[2][3];
      #pragma unroll
      for (int mi = 0; mi < 2; ++mi)
        #pragma unroll
        for (int g = 0; g < 3; ++g) acc[mi][g] = (v8f){};

      for (int k0 = 0; k0 < kH; k0 += 32) {
        const v16bf br = load_b_frag(whh1b, hcol,          kH, k0, lane);
        const v16bf bz = load_b_frag(whh1b, hcol + kH,     kH, k0, lane);
        const v16bf bn = load_b_frag(whh1b, hcol + 2 * kH, kH, k0, lane);
        #pragma unroll
        for (int mi = 0; mi < 2; ++mi) {
          const int m0 = (mgrp * 2 + mi) * 16;
          const v16bf a = load_a_frag(s0_in, m0 + (lane & 15), kH, k0, lane);
          acc[mi][0] = wmma_bf16(a, br, acc[mi][0]);
          acc[mi][1] = wmma_bf16(a, bz, acc[mi][1]);
          acc[mi][2] = wmma_bf16(a, bn, acc[mi][2]);
        }
      }
      const float w0r = wcol0[hcol],          bhr = b_hh1[hcol];
      const float w0z = wcol0[hcol + kH],     bhz = b_hh1[hcol + kH];
      const float w0n = wcol0[hcol + 2 * kH], bhn = b_hh1[hcol + 2 * kH];
      #pragma unroll
      for (int mi = 0; mi < 2; ++mi) {
        const int m0 = (mgrp * 2 + mi) * 16;
        #pragma unroll
        for (int r = 0; r < 8; ++r) {
          const int b  = m0 + r + bhalf;
          const float xc = xcur[b];
          const float* gy = giy + b * kG3;
          const float ir = gy[hcol]          + xc * w0r;
          const float iz = gy[hcol + kH]     + xc * w0z;
          const float in_ = gy[hcol + 2*kH]  + xc * w0n;
          const float rg = sigmoid_f(ir + acc[mi][0][r] + bhr);
          const float zg = sigmoid_f(iz + acc[mi][1][r] + bhz);
          const float ng = tanhf(in_ + rg * (acc[mi][2][r] + bhn));
          const float sold = s0f[inb * kB * kH + b * kH + hcol];
          const float snew = (1.0f - zg) * ng + zg * sold;
          s0f[outb * kB * kH + b * kH + hcol] = snew;
          s0b[outb * kB * kH + b * kH + hcol] = f2bf(snew);
        }
      }
    }
    gbar(bar, ep);

    // ---- PHASE 2: s1_new = GRU2(s0_new, s1) : two GEMMs, 6 acc chains ----
    {
      v8f ai[2][3], ah[2][3];
      #pragma unroll
      for (int mi = 0; mi < 2; ++mi)
        #pragma unroll
        for (int g = 0; g < 3; ++g) { ai[mi][g] = (v8f){}; ah[mi][g] = (v8f){}; }

      for (int k0 = 0; k0 < kH; k0 += 32) {
        const v16bf bir = load_b_frag(wih2b, hcol,          kH, k0, lane);
        const v16bf biz = load_b_frag(wih2b, hcol + kH,     kH, k0, lane);
        const v16bf bin = load_b_frag(wih2b, hcol + 2 * kH, kH, k0, lane);
        const v16bf bhr = load_b_frag(whh2b, hcol,          kH, k0, lane);
        const v16bf bhz = load_b_frag(whh2b, hcol + kH,     kH, k0, lane);
        const v16bf bhn = load_b_frag(whh2b, hcol + 2 * kH, kH, k0, lane);
        #pragma unroll
        for (int mi = 0; mi < 2; ++mi) {
          const int m0 = (mgrp * 2 + mi) * 16;
          const v16bf a0 = load_a_frag(s0_new, m0 + (lane & 15), kH, k0, lane);
          const v16bf a1 = load_a_frag(s1_in,  m0 + (lane & 15), kH, k0, lane);
          ai[mi][0] = wmma_bf16(a0, bir, ai[mi][0]);
          ai[mi][1] = wmma_bf16(a0, biz, ai[mi][1]);
          ai[mi][2] = wmma_bf16(a0, bin, ai[mi][2]);
          ah[mi][0] = wmma_bf16(a1, bhr, ah[mi][0]);
          ah[mi][1] = wmma_bf16(a1, bhz, ah[mi][1]);
          ah[mi][2] = wmma_bf16(a1, bhn, ah[mi][2]);
        }
      }
      const float bi_r = b_ih2[hcol],          bh_r = b_hh2[hcol];
      const float bi_z = b_ih2[hcol + kH],     bh_z = b_hh2[hcol + kH];
      const float bi_n = b_ih2[hcol + 2 * kH], bh_n = b_hh2[hcol + 2 * kH];
      #pragma unroll
      for (int mi = 0; mi < 2; ++mi) {
        const int m0 = (mgrp * 2 + mi) * 16;
        #pragma unroll
        for (int r = 0; r < 8; ++r) {
          const int b = m0 + r + bhalf;
          const float rg = sigmoid_f(ai[mi][0][r] + bi_r + ah[mi][0][r] + bh_r);
          const float zg = sigmoid_f(ai[mi][1][r] + bi_z + ah[mi][1][r] + bh_z);
          const float ng = tanhf(ai[mi][2][r] + bi_n + rg * (ah[mi][2][r] + bh_n));
          const float sold = s1f[inb * kB * kH + b * kH + hcol];
          const float snew = (1.0f - zg) * ng + zg * sold;
          s1f[outb * kB * kH + b * kH + hcol] = snew;
          s1b[outb * kB * kH + b * kH + hcol] = f2bf(snew);
        }
      }
    }
    gbar(bar, ep);

    // ---- PHASE 3: out[:,t] = relu(s1_new @ w_fc^T + b_fc); feeds x_{t+1} ----
    {
      const float* sr = s1f + outb * kB * kH + gw * kH;   // one wave per batch row
      float acc = 0.0f;
      #pragma unroll
      for (int j = 0; j < 16; ++j) acc = fmaf(sr[lane + 32 * j], w_fc[lane + 32 * j], acc);
      #pragma unroll
      for (int off = 16; off > 0; off >>= 1) acc += __shfl_xor(acc, off, 32);
      if (lane == 0) {
        const float v = fmaxf(acc + b_fc[0], 0.0f);
        xcur[gw] = v;
        out[gw * L + t] = v;
      }
    }
    gbar(bar, ep);
  }
}

// ---------------------------------------------------------------------------
extern "C" void kernel_launch(void* const* d_in, const int* in_sizes, int n_in,
                              void* d_out, int out_size, void* d_ws, size_t ws_size,
                              hipStream_t stream) {
  (void)in_sizes; (void)n_in; (void)out_size; (void)ws_size;
  auto F = [&](int i) { return (const float*)d_in[i]; };

  char* p = (char*)d_ws;
  auto carve = [&](size_t bytes) -> char* {
    char* r = p; p += (bytes + 255) & ~(size_t)255; return r;
  };
  int*            bar   = (int*)carve(256);
  float*          xcur  = (float*)carve(kB * sizeof(float));
  float*          s0f   = (float*)carve(2ull * kB * kH * sizeof(float));
  float*          s1f   = (float*)carve(2ull * kB * kH * sizeof(float));
  unsigned short* s0b   = (unsigned short*)carve(2ull * kB * kH * 2);
  unsigned short* s1b   = (unsigned short*)carve(2ull * kB * kH * 2);
  float*          giy   = (float*)carve((size_t)kB * kG3 * sizeof(float));
  unsigned short* whh1b = (unsigned short*)carve((size_t)kG3 * kH * 2);
  unsigned short* wih2b = (unsigned short*)carve((size_t)kG3 * kH * 2);
  unsigned short* whh2b = (unsigned short*)carve((size_t)kG3 * kH * 2);
  float*          wcol0 = (float*)carve(kG3 * sizeof(float));

  init_barrier_kernel<<<1, 1, 0, stream>>>(bar);
  gru_persistent<<<kNWG, kTPB, 0, stream>>>(
      F(0), F(1), F(2), F(3), F(4), F(5), F(6), F(7),
      F(8), F(9), F(10), F(11), F(12), F(13),
      (const int*)d_in[14], (float*)d_out,
      bar, xcur, s0f, s1f, s0b, s1b, giy, whh1b, wih2b, whh2b, wcol0);
}
